// Seq2SeqSLM_Transformer_76338748719817
// MI455X (gfx1250) — compile-verified, matrix operand order checked by reading, not run on hardware
//
#include <hip/hip_runtime.h>
#include <hip/hip_bf16.h>
#include <math.h>

#define B 64
#define S_SRC 512
#define T_TRG 64
#define E 128
#define NH 4
#define HD 32
#define NL 2
#define FF 256
#define VOUT 32000
#define BS (B * S_SRC)
#define LN_EPS 1e-5f

typedef _Float16 half_t;
typedef __attribute__((ext_vector_type(16))) _Float16 v16h;
typedef __attribute__((ext_vector_type(8)))  float    v8f;

// ---------------------------------------------------------------------------
// WMMA fragment helpers (CDNA5 16x16x32 f16 layouts, ISA 7.12.2)
// A (16x32, MxK): lanes 0-15 -> M=0..15 with K {0..7,16..23}; lanes 16-31 same M
// with K {8..15,24..31}. VGPR r<4: K=2r,2r+1 (+8 for hi lanes); r>=4: K=16+2(r-4).
// Per-lane halves are 16 contiguous bytes -> compiler merges to global_load_b128.
// C/D (16x16 f32): lane&15 = N, row = r + 8*(lane>>4).
// B (32x16, KxN) mirror of A: lane&15 = N, K striping as A.
// ---------------------------------------------------------------------------
__device__ __forceinline__ int frag_kbase(int r, int ksel) {
    int kb = (r < 4) ? (2 * r) : (16 + 2 * (r - 4));
    return kb + ksel * 8;
}

__device__ __forceinline__ v16h load_a(const half_t* Ap, int lda, int m0, int k0) {
    int lane = threadIdx.x & 31;
    int m = m0 + (lane & 15);
    int ksel = lane >> 4;
    v16h f;
#pragma unroll
    for (int r = 0; r < 8; ++r) {
        int k = k0 + frag_kbase(r, ksel);
        f[2 * r]     = Ap[(long)m * lda + k];
        f[2 * r + 1] = Ap[(long)m * lda + k + 1];
    }
    return f;
}

// B tile where weights are row-major W[N][K]: b[k][n] = W[n0+n][k0+k]
__device__ __forceinline__ v16h load_b_wt(const half_t* Wp, int ldw, int n0, int k0) {
    int lane = threadIdx.x & 31;
    int n = n0 + (lane & 15);
    int ksel = lane >> 4;
    v16h f;
#pragma unroll
    for (int r = 0; r < 8; ++r) {
        int k = k0 + frag_kbase(r, ksel);
        f[2 * r]     = Wp[(long)n * ldw + k];
        f[2 * r + 1] = Wp[(long)n * ldw + k + 1];
    }
    return f;
}

// B tile from row-major Bm[K][N] at (k0, n0)
__device__ __forceinline__ v16h load_b_rm(const half_t* Bm, int ldb, int k0, int n0) {
    int lane = threadIdx.x & 31;
    int n = n0 + (lane & 15);
    int ksel = lane >> 4;
    v16h f;
#pragma unroll
    for (int r = 0; r < 8; ++r) {
        int k = k0 + frag_kbase(r, ksel);
        f[2 * r]     = Bm[(long)k * ldb + n];
        f[2 * r + 1] = Bm[(long)(k + 1) * ldb + n];
    }
    return f;
}

__device__ __forceinline__ v8f wmma16(v16h a, v16h b, v8f c) {
    return __builtin_amdgcn_wmma_f32_16x16x32_f16(false, a, false, b, (short)0, c,
                                                  false, false);
}

__device__ __forceinline__ float sigm(float x) { return 1.f / (1.f + __expf(-x)); }

// ---------------------------------------------------------------------------
// Generic WMMA GEMM, register-blocked 16x64 per wave (4 accumulators share one
// A fragment per k-step): C[M,N] = A[M,K](f16) * W[N,K]^T(f16) + bias[N]
// out_mode 0: row-major f32 (Cf) and/or f16 (Ch)
// out_mode 1: logits remap  m=t*64+b  ->  Cf[(b*(T-1)+t)*N + n]
// out_mode 2: qkv head pack n -> {which,h,d}, m -> {b,s}; f16 only
// Requires M%16==0, N%64==0, K%32==0.
// ---------------------------------------------------------------------------
__device__ __forceinline__ void store_tile(v8f acc, int m0, int nsub, int N,
                                           const float* __restrict__ bias,
                                           float* __restrict__ Cf,
                                           half_t* __restrict__ Ch, int relu,
                                           int out_mode) {
    int lane = threadIdx.x & 31;
    int nn = nsub + (lane & 15);
    int g = lane >> 4;
    float bv = bias ? bias[nn] : 0.f;
#pragma unroll
    for (int r = 0; r < 8; ++r) {
        int mm = m0 + r + 8 * g;
        float v = acc[r] + bv;
        if (relu) v = fmaxf(v, 0.f);
        if (out_mode == 0) {
            long idx = (long)mm * N + nn;
            if (Cf) Cf[idx] = v;
            if (Ch) Ch[idx] = (half_t)v;
        } else if (out_mode == 1) {
            int t = mm >> 6, bb = mm & 63;  // m = t*64 + b
            Cf[((long)(bb * (T_TRG - 1) + t)) * N + nn] = v;
        } else {  // qkv pack: [which][b][h][s][d] f16
            int bb = mm >> 9, ss = mm & 511;
            int which = nn >> 7, hh = (nn >> 5) & (NH - 1), d = nn & (HD - 1);
            long idx = (long)which * (B * NH * S_SRC * HD) +
                       (((long)(bb * NH + hh)) * S_SRC + ss) * HD + d;
            Ch[idx] = (half_t)v;
        }
    }
}

__global__ void gemm_wmma_kernel(const half_t* __restrict__ A,
                                 const half_t* __restrict__ W,
                                 const float* __restrict__ bias,
                                 float* __restrict__ Cf, half_t* __restrict__ Ch,
                                 int M, int N, int K, int relu, int out_mode) {
    int wid = blockIdx.x * (blockDim.x >> 5) + (threadIdx.x >> 5);
    int nblk = N >> 6;  // 64-wide N blocks
    int total = (M >> 4) * nblk;
    if (wid >= total) return;
    int tm = wid / nblk, tb = wid % nblk;
    int m0 = tm << 4, n0 = tb << 6;

    v8f acc0 = {}, acc1 = {}, acc2 = {}, acc3 = {};
    for (int k0 = 0; k0 < K; k0 += 32) {
        // Unconditional speculative prefetch of next k-step's weight lines
        // (ISA: speculative prefetch drops on translation fail, no counters;
        // one step past the row end still lands inside the workspace arena).
        {
            int lane = threadIdx.x & 31;
            const half_t* pf = W + (long)(n0 + (lane & 15)) * K + (k0 + 32);
            __builtin_prefetch((const void*)pf, 0, 3);
            __builtin_prefetch((const void*)(pf + 32 * K), 0, 3);
        }
        v16h a  = load_a(A, K, m0, k0);
        v16h b0 = load_b_wt(W, K, n0, k0);
        v16h b1 = load_b_wt(W, K, n0 + 16, k0);
        v16h b2 = load_b_wt(W, K, n0 + 32, k0);
        v16h b3 = load_b_wt(W, K, n0 + 48, k0);
        acc0 = wmma16(a, b0, acc0);
        acc1 = wmma16(a, b1, acc1);
        acc2 = wmma16(a, b2, acc2);
        acc3 = wmma16(a, b3, acc3);
    }
    store_tile(acc0, m0, n0, N, bias, Cf, Ch, relu, out_mode);
    store_tile(acc1, m0, n0 + 16, N, bias, Cf, Ch, relu, out_mode);
    store_tile(acc2, m0, n0 + 32, N, bias, Cf, Ch, relu, out_mode);
    store_tile(acc3, m0, n0 + 48, N, bias, Cf, Ch, relu, out_mode);
}

// ---------------------------------------------------------------------------
// Fused flash attention: one block per (b,h); 8 waves; each wave owns 16-row
// Q tiles. Online softmax; P re-fed to WMMA via per-wave LDS staging.
// ---------------------------------------------------------------------------
__global__ void attention_kernel(const half_t* __restrict__ q,
                                 const half_t* __restrict__ k,
                                 const half_t* __restrict__ v,
                                 half_t* __restrict__ ctx) {
    __shared__ half_t Pst[8][16 * 32];
    int wv = threadIdx.x >> 5, lane = threadIdx.x & 31;
    int bh = blockIdx.x;
    const half_t* qb = q + (long)bh * S_SRC * HD;
    const half_t* kb = k + (long)bh * S_SRC * HD;
    const half_t* vb = v + (long)bh * S_SRC * HD;
    int bIdx = bh / NH, hIdx = bh % NH;
    const float scale = 0.1767766952966369f;  // 1/sqrt(32)

    for (int rt = wv; rt < S_SRC / 16; rt += 8) {
        v16h qa = load_a(qb, HD, rt * 16, 0);
        v8f acc0 = {}, acc1 = {};
        float mrow[8], lrow[8];
#pragma unroll
        for (int r = 0; r < 8; ++r) { mrow[r] = -1e30f; lrow[r] = 0.f; }

        for (int c0 = 0; c0 < S_SRC; c0 += 32) {
            v16h bk0 = load_b_wt(kb, HD, c0, 0);  // K tile, transposed
            v16h bk1 = load_b_wt(kb, HD, c0 + 16, 0);
            v8f z = {};
            v8f s0 = wmma16(qa, bk0, z);
            v8f s1 = wmma16(qa, bk1, z);
#pragma unroll
            for (int r = 0; r < 8; ++r) {
                float a0 = s0[r] * scale, a1 = s1[r] * scale;
                float mx = fmaxf(a0, a1);
                mx = fmaxf(mx, __shfl_xor(mx, 1));
                mx = fmaxf(mx, __shfl_xor(mx, 2));
                mx = fmaxf(mx, __shfl_xor(mx, 4));
                mx = fmaxf(mx, __shfl_xor(mx, 8));  // row max within 16-lane group
                float mnew = fmaxf(mrow[r], mx);
                float alpha = __expf(mrow[r] - mnew);
                float p0 = __expf(a0 - mnew), p1 = __expf(a1 - mnew);
                float ps = p0 + p1;
                ps += __shfl_xor(ps, 1);
                ps += __shfl_xor(ps, 2);
                ps += __shfl_xor(ps, 4);
                ps += __shfl_xor(ps, 8);
                lrow[r] = lrow[r] * alpha + ps;
                mrow[r] = mnew;
                acc0[r] *= alpha;
                acc1[r] *= alpha;
                int row = r + 8 * (lane >> 4);
                Pst[wv][row * 32 + (lane & 15)]      = (half_t)p0;
                Pst[wv][row * 32 + 16 + (lane & 15)] = (half_t)p1;
            }
            asm volatile("s_wait_dscnt 0" ::: "memory");  // P stores -> reload
            v16h pa  = load_a(&Pst[wv][0], 32, 0, 0);
            v16h bv0 = load_b_rm(vb, HD, c0, 0);
            v16h bv1 = load_b_rm(vb, HD, c0, 16);
            acc0 = wmma16(pa, bv0, acc0);
            acc1 = wmma16(pa, bv1, acc1);
        }
#pragma unroll
        for (int r = 0; r < 8; ++r) {
            int row = r + 8 * (lane >> 4);
            int s = rt * 16 + row;
            float inv = 1.f / lrow[r];
            int e0 = hIdx * HD + (lane & 15);
            long base = ((long)bIdx * S_SRC + s) * E;
            ctx[base + e0]      = (half_t)(acc0[r] * inv);
            ctx[base + e0 + 16] = (half_t)(acc1[r] * inv);
        }
    }
}

// ---------------------------------------------------------------------------
// Persistent single-workgroup LSTM: 32 waves, state in LDS, gates via WMMA.
// A = [emb_t || h_{t-1}] (64x256 f16); W = [w_ih || w_hh] (512x256 f16).
// ---------------------------------------------------------------------------
__global__ void lstm_kernel(const half_t* __restrict__ embs,  // [T-1][B][E]
                            const half_t* __restrict__ wg,    // [4E][2E]
                            const float* __restrict__ b_ih,
                            const float* __restrict__ b_hh,
                            const float* __restrict__ h0,
                            half_t* __restrict__ hseq) {      // [T-1][B][E]
    extern __shared__ char smem[];
    half_t* Ash = (half_t*)smem;                               // 64*256 f16
    float* Gsh = (float*)(smem + B * 256 * 2);                 // 64*512 f32
    float* Hsh = (float*)(smem + B * 256 * 2 + B * 512 * 4);   // 64*128 f32
    float* Csh = Hsh + B * E;                                  // 64*128 f32
    int tid = threadIdx.x;
    for (int i = tid; i < B * E; i += blockDim.x) { Hsh[i] = h0[i]; Csh[i] = 0.f; }
    __syncthreads();
    int wv = tid >> 5, lane = tid & 31;

    for (int t = 0; t < T_TRG - 1; ++t) {
        for (int i = tid; i < B * 256; i += blockDim.x) {
            int bb = i >> 8, cc = i & 255;
            Ash[i] = (cc < E) ? embs[((long)t * B + bb) * E + cc]
                              : (half_t)Hsh[bb * E + (cc - E)];
        }
        __syncthreads();
        for (int tile = wv; tile < 128; tile += 32) {  // 4x32 tiles of gates
            int m0 = (tile >> 5) << 4, n0 = (tile & 31) << 4;
            v8f acc = {};
            for (int k0 = 0; k0 < 256; k0 += 32) {
                v16h a = load_a(Ash, 256, m0, k0);
                v16h b = load_b_wt(wg, 256, n0, k0);
                acc = wmma16(a, b, acc);
            }
            int nn = n0 + (lane & 15);
            float bv = b_ih[nn] + b_hh[nn];
#pragma unroll
            for (int r = 0; r < 8; ++r) {
                int mm = m0 + r + 8 * (lane >> 4);
                Gsh[mm * 512 + nn] = acc[r] + bv;
            }
        }
        __syncthreads();
        for (int i = tid; i < B * E; i += blockDim.x) {
            int bb = i >> 7, ee = i & 127;
            float gi = Gsh[bb * 512 + ee];
            float gf = Gsh[bb * 512 + 128 + ee];
            float gg = Gsh[bb * 512 + 256 + ee];
            float go = Gsh[bb * 512 + 384 + ee];
            float c = sigm(gf) * Csh[i] + sigm(gi) * tanhf(gg);
            float h = sigm(go) * tanhf(c);
            Csh[i] = c;
            Hsh[i] = h;
            hseq[(long)t * B * E + i] = (half_t)h;
        }
        __syncthreads();
    }
}

// ---------------------------------------------------------------------------
// Small elementwise / reduction kernels
// ---------------------------------------------------------------------------
__global__ void cvt_f16_kernel(const float* __restrict__ s, half_t* __restrict__ d,
                               int n) {
    int i = blockIdx.x * blockDim.x + threadIdx.x;
    if (i < n) d[i] = (half_t)s[i];
}

__global__ void embed_src_kernel(const int* __restrict__ src,
                                 const float* __restrict__ emb,
                                 float* __restrict__ xf, half_t* __restrict__ xh) {
    int i = blockIdx.x * blockDim.x + threadIdx.x;
    if (i >= BS * E) return;
    int tok = src[i >> 7];
    float v = emb[(long)tok * E + (i & 127)];
    xf[i] = v;
    xh[i] = (half_t)v;
}

__global__ void embed_trg_kernel(const int* __restrict__ trg,
                                 const float* __restrict__ emb,
                                 half_t* __restrict__ out) {
    int i = blockIdx.x * blockDim.x + threadIdx.x;
    if (i >= (T_TRG - 1) * B * E) return;
    int t = i / (B * E);
    int rem = i - t * (B * E);
    int bb = rem >> 7, ee = rem & 127;
    int tok = trg[bb * T_TRG + t];
    out[i] = (half_t)emb[(long)tok * E + ee];
}

__global__ void gates_weight_kernel(const float* __restrict__ w_ih,
                                    const float* __restrict__ w_hh,
                                    half_t* __restrict__ wg) {
    int i = blockIdx.x * blockDim.x + threadIdx.x;  // [512][256]
    if (i >= 4 * E * 2 * E) return;
    int n = i >> 8, kk = i & 255;
    float v = (kk < E) ? w_ih[n * E + kk] : w_hh[n * E + (kk - E)];
    wg[i] = (half_t)v;
}

// One wave per row: y = LN(x + dlt) * w + b; writes f32 + f16
__global__ void add_ln_kernel(const float* __restrict__ x,
                              const float* __restrict__ dlt,
                              const float* __restrict__ w,
                              const float* __restrict__ bp,
                              float* __restrict__ yf, half_t* __restrict__ yh) {
    int row = blockIdx.x * (blockDim.x >> 5) + (threadIdx.x >> 5);
    int lane = threadIdx.x & 31;
    if (row >= BS) return;
    float v[4], s = 0.f, ss = 0.f;
#pragma unroll
    for (int j = 0; j < 4; ++j) {
        int e = lane * 4 + j;
        v[j] = x[(long)row * E + e] + dlt[(long)row * E + e];
        s += v[j];
        ss += v[j] * v[j];
    }
#pragma unroll
    for (int m = 1; m < 32; m <<= 1) {
        s += __shfl_xor(s, m);
        ss += __shfl_xor(ss, m);
    }
    float mu = s * (1.f / E);
    float var = ss * (1.f / E) - mu * mu;
    float rstd = rsqrtf(var + LN_EPS);
#pragma unroll
    for (int j = 0; j < 4; ++j) {
        int e = lane * 4 + j;
        float y = (v[j] - mu) * rstd * w[e] + bp[e];
        yf[(long)row * E + e] = y;
        yh[(long)row * E + e] = (half_t)y;
    }
}

__global__ void mean_kernel(const float* __restrict__ x, float* __restrict__ h0) {
    int i = blockIdx.x * blockDim.x + threadIdx.x;  // [B][E]
    if (i >= B * E) return;
    int bb = i >> 7, ee = i & 127;
    float s = 0.f;
    for (int ss = 0; ss < S_SRC; ++ss) s += x[((long)bb * S_SRC + ss) * E + ee];
    h0[i] = s * (1.f / S_SRC);
}

// ---------------------------------------------------------------------------
extern "C" void kernel_launch(void* const* d_in, const int* in_sizes, int n_in,
                              void* d_out, int out_size, void* d_ws, size_t ws_size,
                              hipStream_t stream) {
    const int* src = (const int*)d_in[0];
    const int* trg = (const int*)d_in[1];
    const float* emb_in = (const float*)d_in[2];
    const float* emb_out = (const float*)d_in[3];
    const float* wqkv = (const float*)d_in[4];
    const float* bqkv = (const float*)d_in[5];
    const float* wo = (const float*)d_in[6];
    const float* bo = (const float*)d_in[7];
    const float* ln1w = (const float*)d_in[8];
    const float* ln1b = (const float*)d_in[9];
    const float* w1 = (const float*)d_in[10];
    const float* b1 = (const float*)d_in[11];
    const float* w2 = (const float*)d_in[12];
    const float* b2 = (const float*)d_in[13];
    const float* ln2w = (const float*)d_in[14];
    const float* ln2b = (const float*)d_in[15];
    const float* w_ih = (const float*)d_in[16];
    const float* w_hh = (const float*)d_in[17];
    const float* b_ih = (const float*)d_in[18];
    const float* b_hh = (const float*)d_in[19];
    const float* fc_w = (const float*)d_in[20];
    const float* fc_b = (const float*)d_in[21];

    char* ws = (char*)d_ws;
    size_t o = 0;
    float* x_f32 = (float*)(ws + o);      o += (size_t)BS * E * 4;
    half_t* x_h = (half_t*)(ws + o);      o += (size_t)BS * E * 2;
    half_t* qkv_h = (half_t*)(ws + o);    o += (size_t)3 * B * NH * S_SRC * HD * 2;
    half_t* ff1_h = qkv_h;  // reused after attention consumes qkv
    half_t* ctx_h = (half_t*)(ws + o);    o += (size_t)BS * E * 2;
    float* tmp_f32 = (float*)(ws + o);    o += (size_t)BS * E * 4;
    half_t* wqkv_h = (half_t*)(ws + o);   o += (size_t)NL * 3 * E * E * 2;
    half_t* wo_h = (half_t*)(ws + o);     o += (size_t)NL * E * E * 2;
    half_t* w1_h = (half_t*)(ws + o);     o += (size_t)NL * FF * E * 2;
    half_t* w2_h = (half_t*)(ws + o);     o += (size_t)NL * E * FF * 2;
    half_t* wg_h = (half_t*)(ws + o);     o += (size_t)4 * E * 2 * E * 2;
    half_t* fcw_h = (half_t*)(ws + o);    o += (size_t)VOUT * E * 2;
    half_t* embs_h = (half_t*)(ws + o);   o += (size_t)(T_TRG - 1) * B * E * 2;
    half_t* hseq_h = (half_t*)(ws + o);   o += (size_t)(T_TRG - 1) * B * E * 2;
    float* h0 = (float*)(ws + o);         o += (size_t)B * E * 4;

    auto cvt = [&](const float* s, half_t* d, int n) {
        cvt_f16_kernel<<<(n + 255) / 256, 256, 0, stream>>>(s, d, n);
    };
    auto gemm = [&](const half_t* A, const half_t* W, const float* bias, float* Cf,
                    half_t* Ch, int M, int N, int K, int relu, int mode) {
        int waves = (M / 16) * (N / 64);  // one 16x64 block per wave
        gemm_wmma_kernel<<<(waves + 7) / 8, 256, 0, stream>>>(A, W, bias, Cf, Ch, M,
                                                              N, K, relu, mode);
    };

    // weight conversions
    cvt(wqkv, wqkv_h, NL * 3 * E * E);
    cvt(wo, wo_h, NL * E * E);
    cvt(w1, w1_h, NL * FF * E);
    cvt(w2, w2_h, NL * E * FF);
    cvt(fc_w, fcw_h, VOUT * E);
    gates_weight_kernel<<<(4 * E * 2 * E + 255) / 256, 256, 0, stream>>>(w_ih, w_hh,
                                                                         wg_h);
    // embeddings
    embed_src_kernel<<<(BS * E + 255) / 256, 256, 0, stream>>>(src, emb_in, x_f32,
                                                               x_h);
    embed_trg_kernel<<<((T_TRG - 1) * B * E + 255) / 256, 256, 0, stream>>>(
        trg, emb_out, embs_h);

    const int qkvStride = B * NH * S_SRC * HD;
    for (int l = 0; l < NL; ++l) {
        gemm(x_h, wqkv_h + (size_t)l * 3 * E * E, bqkv + l * 3 * E, nullptr, qkv_h,
             BS, 3 * E, E, 0, 2);
        attention_kernel<<<B * NH, 256, 0, stream>>>(qkv_h, qkv_h + qkvStride,
                                                     qkv_h + 2 * qkvStride, ctx_h);
        gemm(ctx_h, wo_h + (size_t)l * E * E, bo + l * E, tmp_f32, nullptr, BS, E, E,
             0, 0);
        add_ln_kernel<<<BS / 8, 256, 0, stream>>>(x_f32, tmp_f32, ln1w + l * E,
                                                  ln1b + l * E, x_f32, x_h);
        gemm(x_h, w1_h + (size_t)l * FF * E, b1 + l * FF, nullptr, ff1_h, BS, FF, E,
             1, 0);
        gemm(ff1_h, w2_h + (size_t)l * E * FF, b2 + l * E, tmp_f32, nullptr, BS, E,
             FF, 0, 0);
        add_ln_kernel<<<BS / 8, 256, 0, stream>>>(x_f32, tmp_f32, ln2w + l * E,
                                                  ln2b + l * E, x_f32, x_h);
    }

    mean_kernel<<<(B * E + 255) / 256, 256, 0, stream>>>(x_f32, h0);

    // Persistent LSTM: 1 workgroup, 32 waves, ~224KB dynamic LDS (<=320KB/WGP)
    size_t lstm_lds =
        (size_t)B * 256 * 2 + (size_t)B * 512 * 4 + 2 * (size_t)B * E * 4;
    lstm_kernel<<<1, 1024, lstm_lds, stream>>>(embs_h, wg_h, b_ih, b_hh, h0, hseq_h);

    // Batched logits: [4032 x 32000 x 128], remapped to [B][T-1][V]
    gemm(hseq_h, fcw_h, fc_b, (float*)d_out, nullptr, (T_TRG - 1) * B, VOUT, E, 0, 1);
}
